// SkipgramModel_81458349736088
// MI455X (gfx1250) — matrix operand order repaired and checked by reading, not compiled, not run on hardware
//
#include <hip/hip_runtime.h>
#include <hip/hip_bf16.h>
#include <hip/hip_fp16.h>

typedef __attribute__((ext_vector_type(16))) _Float16 v16h;
typedef __attribute__((ext_vector_type(8)))  _Float16 v8h;
typedef __attribute__((ext_vector_type(8)))  float    v8f;

constexpr int BATCH = 8192;
constexpr int K     = 300;     // real K
constexpr int KP    = 320;     // K padded to multiple of 32 (WMMA K)
constexpr int N     = 50257;   // pred vocab
constexpr int KS    = 328;     // LDS K-stride in halves (bank-conflict padding)
constexpr int BM    = 128;
constexpr int BN    = 128;
constexpr int NBLK  = (N + BN - 1) / BN;   // 393
constexpr int NPAD  = NBLK * BN;           // 50304 (zero-padded cols)
constexpr int MBLK  = BATCH / BM;          // 64
constexpr size_t SMEM_BYTES = (size_t)(BM + BN) * KS * sizeof(_Float16); // 167936

// ---------------------------------------------------------------------------
// Pass 1: gather embedding rows of W1^T (columns of W1) -> f16, K padded to 320
// ---------------------------------------------------------------------------
__global__ __launch_bounds__(256) void embed_gather(const int* __restrict__ x,
                                                    const float* __restrict__ W1,
                                                    _Float16* __restrict__ embh) {
  const int idx = blockIdx.x * 256 + threadIdx.x;   // exact: BATCH*KP = 2621440
  const int b = idx / KP;
  const int k = idx - b * KP;
  float v = 0.f;
  if (k < K) v = W1[(size_t)k * N + x[b]];
  embh[idx] = (_Float16)v;
}

// ---------------------------------------------------------------------------
// Pass 2: W2 fp32 -> f16, padded to [NPAD][KP] with zeros (one-time, L2-resident)
// ---------------------------------------------------------------------------
__global__ __launch_bounds__(256) void w2_convert(const float* __restrict__ W2,
                                                  _Float16* __restrict__ w2h) {
  const int idx = blockIdx.x * 256 + threadIdx.x;   // exact: NPAD*KP = 16097280
  const int n = idx / KP;
  const int k = idx - n * KP;
  float v = 0.f;
  if (n < N && k < K) v = W2[(size_t)n * K + k];
  w2h[idx] = (_Float16)v;
}

// ---------------------------------------------------------------------------
// GEMM core: 128x128 tile, K fully LDS-resident, 8 wave32s x 8 WMMA tiles.
//   WRITE=false : accumulate per-row sum(exp(logit)) partials -> ws
//   WRITE=true  : recompute logits, store logit - lse[row] -> out
// ---------------------------------------------------------------------------
template <bool WRITE>
__global__ __launch_bounds__(256) void gemm_logits(const _Float16* __restrict__ embh,
                                                   const _Float16* __restrict__ w2h,
                                                   float* __restrict__ out,
                                                   float* __restrict__ partial,
                                                   const float* __restrict__ lse) {
  extern __shared__ char smem[];
  _Float16* Alds = (_Float16*)smem;            // [BM][KS]
  _Float16* Blds = Alds + BM * KS;             // [BN][KS]

  const int tid = threadIdx.x;
  const int nb = blockIdx.x, mb = blockIdx.y;
  const int rowBase = mb * BM;
  const int colBase = nb * BN;

  // ---- stage A tile: 128 rows x 320 halves ----
  for (int i = tid; i < BM * (KP / 8); i += 256) {
    const int r = i / (KP / 8);
    const int seg = i - r * (KP / 8);
    const uint4 v = *(const uint4*)(embh + (size_t)(rowBase + r) * KP + seg * 8);
    *(uint4*)(Alds + r * KS + seg * 8) = v;
  }
  // ---- stage B tile: 128 cols x 320 halves (w2h is pre-padded, no checks) ----
  for (int i = tid; i < BN * (KP / 8); i += 256) {
    const int c = i / (KP / 8);
    const int seg = i - c * (KP / 8);
    const uint4 v = *(const uint4*)(w2h + (size_t)(colBase + c) * KP + seg * 8);
    *(uint4*)(Blds + c * KS + seg * 8) = v;
  }
  __syncthreads();

  const int wid  = tid >> 5;
  const int lane = tid & 31;
  const int wr   = wid >> 2;      // 0..1 : wave row  (64 rows each)
  const int wc   = wid & 3;       // 0..3 : wave col  (32 cols each)
  const int r16  = lane & 15;
  const int hh   = lane >> 4;     // half-wave select

  v8f acc[4][2] = {};

#pragma unroll
  for (int kk = 0; kk < KP / 32; ++kk) {
    const int kb = kk * 32;
    // A fragments (16x32 f16): lane(r16,hh) -> K = kb+8*hh..+7 and kb+16+8*hh..+7
    v16h af[4];
#pragma unroll
    for (int mt = 0; mt < 4; ++mt) {
      const _Float16* p = Alds + (wr * 64 + mt * 16 + r16) * KS + kb + hh * 8;
      const v8h lo = *(const v8h*)p;
      const v8h hi = *(const v8h*)(p + 16);
      af[mt] = __builtin_shufflevector(lo, hi, 0,1,2,3,4,5,6,7,8,9,10,11,12,13,14,15);
    }
    // B fragments (32x16 f16): lane(r16,hh) -> col r16, K = kb+8*hh..+7 and kb+16+8*hh..+7
    v16h bf[2];
#pragma unroll
    for (int nt = 0; nt < 2; ++nt) {
      const _Float16* p = Blds + (wc * 32 + nt * 16 + r16) * KS + kb + hh * 8;
      const v8h lo = *(const v8h*)p;
      const v8h hi = *(const v8h*)(p + 16);
      bf[nt] = __builtin_shufflevector(lo, hi, 0,1,2,3,4,5,6,7,8,9,10,11,12,13,14,15);
    }
#pragma unroll
    for (int mt = 0; mt < 4; ++mt)
#pragma unroll
      for (int nt = 0; nt < 2; ++nt)
        acc[mt][nt] = __builtin_amdgcn_wmma_f32_16x16x32_f16(
            false, af[mt], false, bf[nt], (short)0, acc[mt][nt], false, false);
  }

  __syncthreads();                       // LDS tiles dead; reuse below
  float* fsm = (float*)smem;

  if constexpr (WRITE) {
    // ---- pass B epilogue: out = logit - lse[row] ----
    if (tid < BM) fsm[tid] = lse[rowBase + tid];
    __syncthreads();
#pragma unroll
    for (int mt = 0; mt < 4; ++mt) {
#pragma unroll
      for (int j = 0; j < 8; ++j) {
        const int lrow = wr * 64 + mt * 16 + hh * 8 + j;   // C layout: M = j + 8*hh
        const size_t grow = (size_t)(rowBase + lrow);
        const float l = fsm[lrow];
#pragma unroll
        for (int nt = 0; nt < 2; ++nt) {
          const int col = colBase + wc * 32 + nt * 16 + r16;
          if (col < N) out[grow * (size_t)N + col] = acc[mt][nt][j] - l;
        }
      }
    }
  } else {
    // ---- pass A epilogue: per-row sum(exp) over this 128-col slab ----
#pragma unroll
    for (int mt = 0; mt < 4; ++mt) {
#pragma unroll
      for (int j = 0; j < 8; ++j) {
        const int lrow = wr * 64 + mt * 16 + hh * 8 + j;
        float s = 0.f;
#pragma unroll
        for (int nt = 0; nt < 2; ++nt) {
          const int col = colBase + wc * 32 + nt * 16 + r16;
          if (col < N) s += __expf(acc[mt][nt][j]);   // logits ~ +-0.05: no max-shift
        }
        s += __shfl_xor(s, 8, 32);
        s += __shfl_xor(s, 4, 32);
        s += __shfl_xor(s, 2, 32);
        s += __shfl_xor(s, 1, 32);
        if (r16 == 0) fsm[wc * 128 + lrow] = s;   // deterministic, no atomics
      }
    }
    __syncthreads();
    if (tid < BM) {
      const float s = fsm[tid] + fsm[128 + tid] + fsm[256 + tid] + fsm[384 + tid];
      partial[(size_t)nb * BATCH + rowBase + tid] = s;
    }
  }
}

// ---------------------------------------------------------------------------
// lse[b] = log( sum over 393 partials )
// ---------------------------------------------------------------------------
__global__ __launch_bounds__(256) void row_lse(const float* __restrict__ partial,
                                               float* __restrict__ lse) {
  const int r = blockIdx.x * 256 + threadIdx.x;   // exactly BATCH threads
  float s = 0.f;
  for (int nb = 0; nb < NBLK; ++nb) s += partial[(size_t)nb * BATCH + r];
  lse[r] = logf(s);
}

// ---------------------------------------------------------------------------
extern "C" void kernel_launch(void* const* d_in, const int* in_sizes, int n_in,
                              void* d_out, int out_size, void* d_ws, size_t ws_size,
                              hipStream_t stream) {
  const int*   x  = (const int*)d_in[0];
  const float* W1 = (const float*)d_in[1];   // [300][50257]
  const float* W2 = (const float*)d_in[2];   // [50257][300]
  float* out = (float*)d_out;

  // workspace layout (~50.4 MB total)
  char* ws = (char*)d_ws;
  _Float16* embh = (_Float16*)ws;                               //  5,242,880 B
  _Float16* w2h  = (_Float16*)(ws + (size_t)BATCH * KP * 2);    // 32,194,560 B
  float*    part = (float*)((char*)w2h + (size_t)NPAD * KP * 2);// 12,877,824 B
  float*    lse  = (float*)((char*)part + (size_t)NBLK * BATCH * 4); // 32,768 B

  (void)hipFuncSetAttribute((const void*)gemm_logits<false>,
                            hipFuncAttributeMaxDynamicSharedMemorySize,
                            (int)SMEM_BYTES);
  (void)hipFuncSetAttribute((const void*)gemm_logits<true>,
                            hipFuncAttributeMaxDynamicSharedMemorySize,
                            (int)SMEM_BYTES);

  embed_gather<<<(BATCH * KP) / 256, 256, 0, stream>>>(x, W1, embh);
  w2_convert<<<(NPAD * KP) / 256, 256, 0, stream>>>(W2, w2h);
  gemm_logits<false><<<dim3(NBLK, MBLK), 256, SMEM_BYTES, stream>>>(
      embh, w2h, nullptr, part, nullptr);
  row_lse<<<BATCH / 256, 256, 0, stream>>>(part, lse);
  gemm_logits<true><<<dim3(NBLK, MBLK), 256, SMEM_BYTES, stream>>>(
      embh, w2h, out, nullptr, lse);
}
